// SequenceQuantizerEMA_89867895701685
// MI455X (gfx1250) — matrix operand order repaired and checked by reading, not compiled, not run on hardware
//
#include <hip/hip_runtime.h>
#include <hip/hip_bf16.h>

// ---------------- problem constants ----------------
#define BB 64
#define SS 512
#define DD 512
#define KK 1024
#define NN (BB * SS)          // 32768 tokens
#define DECAY 0.99f
#define EPSV 1e-05f
#define COMMIT 0.25f

typedef __attribute__((ext_vector_type(16))) __bf16 bf16x16;
typedef __attribute__((ext_vector_type(4)))  __bf16 bf16x4;
typedef __attribute__((ext_vector_type(8)))  float  f32x8;

// bf16 round-to-nearest-even split helpers (bit ops; no reliance on native cvt)
static __device__ __forceinline__ __bf16 f2bf(float f) {
    unsigned int u = __builtin_bit_cast(unsigned int, f);
    unsigned int r = (u + 0x7FFFu + ((u >> 16) & 1u)) >> 16;
    unsigned short s = (unsigned short)r;
    return __builtin_bit_cast(__bf16, s);
}
static __device__ __forceinline__ float bf2f(__bf16 b) {
    unsigned short s = __builtin_bit_cast(unsigned short, b);
    unsigned int u = ((unsigned int)s) << 16;
    return __builtin_bit_cast(float, u);
}

#define WMMA_BF16(A, B, C) \
    __builtin_amdgcn_wmma_f32_16x16x32_bf16(false, (A), false, (B), (short)0, (C), false, false)

// ---------------- kernel 1: zero scratch ----------------
__global__ void vq_zero_kernel(float* __restrict__ p, int n) {
    int i = blockIdx.x * blockDim.x + threadIdx.x;
    if (i < n) p[i] = 0.0f;
}

// ---------------- kernel 2: codebook -> bf16 hi/lo + wnorm ----------------
__global__ __launch_bounds__(256)
void vq_prep_codebook(const float* __restrict__ w,
                      __bf16* __restrict__ cb_hi, __bf16* __restrict__ cb_lo,
                      float* __restrict__ wnorm) {
    __shared__ float red[256];
    int k = blockIdx.x;
    int tid = threadIdx.x;
    float ss = 0.0f;
    for (int d = tid; d < DD; d += 256) {
        float v = w[(size_t)k * DD + d];
        __bf16 hi = f2bf(v);
        __bf16 lo = f2bf(v - bf2f(hi));
        cb_hi[(size_t)k * DD + d] = hi;
        cb_lo[(size_t)k * DD + d] = lo;
        ss += v * v;
    }
    red[tid] = ss;
    __syncthreads();
    for (int s = 128; s > 0; s >>= 1) {
        if (tid < s) red[tid] += red[tid + s];
        __syncthreads();
    }
    if (tid == 0) wnorm[k] = red[0];
}

// ---------------- kernel 3: WMMA distance + argmin ----------------
// One wave handles 32 tokens x all K codes, 2x2 tiled (32 rows x 32 codes per
// pass): 12 independent-chain WMMAs per k-step vs 8 LDS + 8 global b128 loads.
// dist = ||w||^2 - 2*x.w ; fp32 dot via bf16 split: hi*hi + hi*lo + lo*hi.
__global__ __launch_bounds__(32)
void vq_argmin_kernel(const float* __restrict__ inputs,
                      const unsigned char* __restrict__ pad_mask,
                      const __bf16* __restrict__ cb_hi,
                      const __bf16* __restrict__ cb_lo,
                      const float* __restrict__ wnorm,
                      int* __restrict__ idx_i,
                      float* __restrict__ idx_f,
                      float* __restrict__ counts,
                      float* __restrict__ nonpad_acc) {
    // Two 16-row A-fragment sets in contiguous-k A layout (k = 16h + s). 32 KB + 32 KB.
    __shared__ __align__(32) __bf16 aHi[32 * DD];
    __shared__ __align__(32) __bf16 aLo[32 * DD];

    const int lane = threadIdx.x;
    const int row0 = blockIdx.x * 32;

    // ---- stage A (32 rows x 512 d) as bf16 hi/lo in fragment order ----
    for (int i4 = lane; i4 < 32 * (DD / 4); i4 += 32) {
        int m  = i4 >> 7;          // row 0..31
        int d0 = (i4 & 127) << 2;  // 4-element chunk
        const float4 v = *(const float4*)(inputs + (size_t)(row0 + m) * DD + d0);
        bf16x4 h, l;
        h.x = f2bf(v.x); l.x = f2bf(v.x - bf2f(h.x));
        h.y = f2bf(v.y); l.y = f2bf(v.y - bf2f(h.y));
        h.z = f2bf(v.z); l.z = f2bf(v.z - bf2f(h.z));
        h.w = f2bf(v.w); l.w = f2bf(v.w - bf2f(h.w));
        // A-layout within a 16-row tile: lane<16 holds K(d%32)=0..15 (j0..15);
        // lane>=16 holds 16..31 (j0..15).
        int mt = m >> 4;           // row tile 0/1
        int mm = m & 15;
        int ks = d0 >> 5;
        int dl = d0 & 31;
        int laneT, j;
        if (dl < 16)      { laneT = mm;      j = dl;      }
        else              { laneT = mm + 16; j = dl - 16; }
        int off = ((((mt << 4) + ks) * 32 + laneT) << 4) + j;   // elements
        *(bf16x4*)(aHi + off) = h;
        *(bf16x4*)(aLo + off) = l;
    }
    __syncthreads();

    const int nLane = lane & 15;        // column within 16-wide code tile
    const int dHalf = (lane >> 4) << 4; // B half: K 0..15 vs 16..31

    float bestVal0[8], bestVal1[8];
    int   bestIdx0[8], bestIdx1[8];
#pragma unroll
    for (int v = 0; v < 8; ++v) {
        bestVal0[v] = __builtin_inff(); bestIdx0[v] = 0;
        bestVal1[v] = __builtin_inff(); bestIdx1[v] = 0;
    }

    for (int t = 0; t < KK / 32; ++t) {
        const int code0 = t * 32 + nLane;
        const int code1 = code0 + 16;
        const float wn0 = wnorm[code0];
        const float wn1 = wnorm[code1];
        const __bf16* bh0 = cb_hi + ((size_t)code0 << 9) + dHalf;
        const __bf16* bl0 = cb_lo + ((size_t)code0 << 9) + dHalf;
        const __bf16* bh1 = cb_hi + ((size_t)code1 << 9) + dHalf;
        const __bf16* bl1 = cb_lo + ((size_t)code1 << 9) + dHalf;

        f32x8 a00 = {0.f,0.f,0.f,0.f,0.f,0.f,0.f,0.f};
        f32x8 a01 = a00, a10 = a00, a11 = a00;
#pragma unroll 2
        for (int ks = 0; ks < DD / 32; ++ks) {
            bf16x16 ahi0 = *(const bf16x16*)(aHi + ((ks * 32 + lane) << 4));
            bf16x16 alo0 = *(const bf16x16*)(aLo + ((ks * 32 + lane) << 4));
            bf16x16 ahi1 = *(const bf16x16*)(aHi + (((256 + ks * 32) + lane) << 4));
            bf16x16 alo1 = *(const bf16x16*)(aLo + (((256 + ks * 32) + lane) << 4));
            bf16x16 bhi0 = *(const bf16x16*)(bh0 + ks * 32);
            bf16x16 blo0 = *(const bf16x16*)(bl0 + ks * 32);
            bf16x16 bhi1 = *(const bf16x16*)(bh1 + ks * 32);
            bf16x16 blo1 = *(const bf16x16*)(bl1 + ks * 32);
            a00 = WMMA_BF16(alo0, bhi0, a00);
            a01 = WMMA_BF16(alo0, bhi1, a01);
            a10 = WMMA_BF16(alo1, bhi0, a10);
            a11 = WMMA_BF16(alo1, bhi1, a11);
            a00 = WMMA_BF16(ahi0, blo0, a00);
            a01 = WMMA_BF16(ahi0, blo1, a01);
            a10 = WMMA_BF16(ahi1, blo0, a10);
            a11 = WMMA_BF16(ahi1, blo1, a11);
            a00 = WMMA_BF16(ahi0, bhi0, a00);
            a01 = WMMA_BF16(ahi0, bhi1, a01);
            a10 = WMMA_BF16(ahi1, bhi0, a10);
            a11 = WMMA_BF16(ahi1, bhi1, a11);
        }
        // col tile 0 may contain the banned code 0 (non-pad tokens skip it)
        const bool banned0 = (code0 == 0);
#pragma unroll
        for (int v = 0; v < 8; ++v) {
            float d00 = banned0 ? __builtin_inff() : (wn0 - 2.0f * a00[v]);
            float d10 = banned0 ? __builtin_inff() : (wn0 - 2.0f * a10[v]);
            float d01 = wn1 - 2.0f * a01[v];
            float d11 = wn1 - 2.0f * a11[v];
            if (d00 < bestVal0[v]) { bestVal0[v] = d00; bestIdx0[v] = code0; }
            if (d01 < bestVal0[v]) { bestVal0[v] = d01; bestIdx0[v] = code1; }
            if (d10 < bestVal1[v]) { bestVal1[v] = d10; bestIdx1[v] = code0; }
            if (d11 < bestVal1[v]) { bestVal1[v] = d11; bestIdx1[v] = code1; }
        }
    }

    // reduce over the 16 lanes holding each row's candidate columns
#pragma unroll
    for (int mgrp = 1; mgrp < 16; mgrp <<= 1) {
#pragma unroll
        for (int v = 0; v < 8; ++v) {
            float ov0 = __shfl_xor(bestVal0[v], mgrp, 32);
            int   oi0 = __shfl_xor(bestIdx0[v], mgrp, 32);
            if (ov0 < bestVal0[v] || (ov0 == bestVal0[v] && oi0 < bestIdx0[v])) {
                bestVal0[v] = ov0; bestIdx0[v] = oi0;
            }
            float ov1 = __shfl_xor(bestVal1[v], mgrp, 32);
            int   oi1 = __shfl_xor(bestIdx1[v], mgrp, 32);
            if (ov1 < bestVal1[v] || (ov1 == bestVal1[v] && oi1 < bestIdx1[v])) {
                bestVal1[v] = ov1; bestIdx1[v] = oi1;
            }
        }
    }

    if (nLane == 0) {
        const int half = lane >> 4;
#pragma unroll
        for (int v = 0; v < 8; ++v) {
            // row tile 0
            {
                int row = row0 + v + 8 * half;
                bool pad = pad_mask[row] != 0;
                int id = pad ? 0 : bestIdx0[v];
                idx_i[row] = id;
                idx_f[row] = (float)id;
                atomicAdd(&counts[id], 1.0f);
                if (!pad) atomicAdd(nonpad_acc, 1.0f);
            }
            // row tile 1
            {
                int row = row0 + 16 + v + 8 * half;
                bool pad = pad_mask[row] != 0;
                int id = pad ? 0 : bestIdx1[v];
                idx_i[row] = id;
                idx_f[row] = (float)id;
                atomicAdd(&counts[id], 1.0f);
                if (!pad) atomicAdd(nonpad_acc, 1.0f);
            }
        }
    }
}

// ---------------- kernel 4: gather quantized, loss, dw scatter ----------------
__global__ __launch_bounds__(256)
void vq_gather_kernel(const float* __restrict__ inputs,
                      const float* __restrict__ codebook,
                      const int* __restrict__ idx_i,
                      float* __restrict__ out_q,
                      float* __restrict__ dw,
                      float* __restrict__ loss_acc) {
    __shared__ float red[256];
    int row = blockIdx.x;
    int tid = threadIdx.x;
    int id = idx_i[row];
    float lsum = 0.0f;
    for (int d = tid; d < DD; d += 256) {
        float q = codebook[(size_t)id * DD + d];
        float x = inputs[(size_t)row * DD + d];
        out_q[(size_t)row * DD + d] = q;   // straight-through value == quantized
        float diff = q - x;
        lsum += diff * diff;
        atomicAdd(&dw[(size_t)id * DD + d], x);
    }
    red[tid] = lsum;
    __syncthreads();
    for (int s = 128; s > 0; s >>= 1) {
        if (tid < s) red[tid] += red[tid + s];
        __syncthreads();
    }
    if (tid == 0) atomicAdd(loss_acc, red[0]);
}

// ---------------- kernel 5: EMA cluster sizes, n, entropy ----------------
__global__ __launch_bounds__(256)
void vq_stats_kernel(const float* __restrict__ ema_cs,
                     const float* __restrict__ counts,
                     float* __restrict__ cs_raw,
                     float* __restrict__ n_acc,
                     float* __restrict__ h_acc) {
    __shared__ float redN[256];
    __shared__ float redH[256];
    int tid = threadIdx.x;
    int k = blockIdx.x * 256 + tid;
    float c = counts[k];
    float cs = ema_cs[k] * DECAY + (1.0f - DECAY) * c;
    cs_raw[k] = cs;
    float ap = c / (float)NN;
    float h = ap * __logf(ap + 1e-10f);
    redN[tid] = cs;
    redH[tid] = h;
    __syncthreads();
    for (int s = 128; s > 0; s >>= 1) {
        if (tid < s) { redN[tid] += redN[tid + s]; redH[tid] += redH[tid + s]; }
        __syncthreads();
    }
    if (tid == 0) {
        atomicAdd(n_acc, redN[0]);
        atomicAdd(h_acc, redH[0]);
    }
}

// ---------------- kernel 6: finalize outputs ----------------
__global__ __launch_bounds__(256)
void vq_final_kernel(const float* __restrict__ ema_w,
                     const float* __restrict__ dw,
                     const float* __restrict__ cs_raw,
                     const float* __restrict__ scalars, // [0]=loss_sum [1]=nonpad [2]=n [3]=H
                     float* __restrict__ out_cs,
                     float* __restrict__ out_ema,
                     float* __restrict__ out_cb,
                     float* __restrict__ out_loss,
                     float* __restrict__ out_ppl) {
    size_t i = (size_t)blockIdx.x * 256 + threadIdx.x;   // over K*D
    int k = (int)(i >> 9);
    float n = scalars[2];
    float cs = (cs_raw[k] + EPSV) / (n + (float)KK * EPSV) * n;
    float ne = ema_w[i] * DECAY + (1.0f - DECAY) * dw[i];
    out_ema[i] = ne;
    out_cb[i] = (k == 0) ? 0.0f : (ne / cs);
    if ((i & 511) == 0) out_cs[k] = cs;
    if (i == 0) {
        float nonpad = scalars[1];
        out_loss[0] = COMMIT * scalars[0] / (nonpad * (float)DD);
        out_ppl[0] = __expf(-scalars[3]);
    }
}

// ---------------- launch ----------------
extern "C" void kernel_launch(void* const* d_in, const int* in_sizes, int n_in,
                              void* d_out, int out_size, void* d_ws, size_t ws_size,
                              hipStream_t stream) {
    const float*         inputs   = (const float*)d_in[0];
    const unsigned char* pad_mask = (const unsigned char*)d_in[1]; // jax bool = 1B
    const float*         codebook = (const float*)d_in[2];
    const float*         ema_cs   = (const float*)d_in[3];
    const float*         ema_w    = (const float*)d_in[4];

    char* ws = (char*)d_ws;
    float*  wnorm   = (float*)(ws + 0);          // K
    float*  counts  = (float*)(ws + 4096);       // K
    float*  scalars = (float*)(ws + 8192);       // 16 floats
    float*  cs_raw  = (float*)(ws + 8448);       // K
    int*    idx_i   = (int*)  (ws + 12544);      // N
    __bf16* cb_hi   = (__bf16*)(ws + 143616);    // K*D
    __bf16* cb_lo   = (__bf16*)(ws + 1192192);   // K*D
    float*  dw      = (float*)(ws + 2240768);    // K*D

    float* out      = (float*)d_out;
    float* out_q    = out;                           // N*D
    float* out_idx  = out + (size_t)NN * DD;         // N
    float* out_loss = out_idx + NN;                  // 1
    float* out_ppl  = out_loss + 1;                  // 1
    float* out_cs   = out_ppl + 1;                   // K
    float* out_ema  = out_cs + KK;                   // K*D
    float* out_cb   = out_ema + (size_t)KK * DD;     // K*D

    // 1. zero scratch accumulators
    vq_zero_kernel<<<(KK + 255) / 256, 256, 0, stream>>>(counts, KK);
    vq_zero_kernel<<<1, 256, 0, stream>>>(scalars, 16);
    vq_zero_kernel<<<(KK * DD + 255) / 256, 256, 0, stream>>>(dw, KK * DD);

    // 2. codebook prep
    vq_prep_codebook<<<KK, 256, 0, stream>>>(codebook, cb_hi, cb_lo, wnorm);

    // 3. WMMA distances + argmin (one wave per 32 tokens, 2x2 tiled)
    vq_argmin_kernel<<<NN / 32, 32, 0, stream>>>(inputs, pad_mask, cb_hi, cb_lo,
                                                 wnorm, idx_i, out_idx,
                                                 counts, &scalars[1]);

    // 4. gather + loss + dw scatter
    vq_gather_kernel<<<NN, 256, 0, stream>>>(inputs, codebook, idx_i,
                                             out_q, dw, &scalars[0]);

    // 5. EMA cluster-size stats
    vq_stats_kernel<<<KK / 256, 256, 0, stream>>>(ema_cs, counts, cs_raw,
                                                  &scalars[2], &scalars[3]);

    // 6. finalize
    vq_final_kernel<<<(KK * DD) / 256, 256, 0, stream>>>(ema_w, dw, cs_raw, scalars,
                                                         out_cs, out_ema, out_cb,
                                                         out_loss, out_ppl);
}
